// MultiHeadSelfAttention_57097295233305
// MI455X (gfx1250) — matrix operand order repaired and checked by reading, not compiled, not run on hardware
//
#include <hip/hip_runtime.h>

#define N_TOK 4096
#define C_DIM 512
#define H_NUM 8
#define D_HEAD 64

typedef _Float16 half_t;
typedef __attribute__((ext_vector_type(8)))  _Float16 v8h;
typedef __attribute__((ext_vector_type(16))) _Float16 v16h;
typedef __attribute__((ext_vector_type(8)))  float    v8f;

static __device__ __forceinline__ v16h hcat(v8h lo, v8h hi) {
  v16h r;
#pragma unroll
  for (int i = 0; i < 8; ++i) { r[i] = lo[i]; r[i + 8] = hi[i]; }
  return r;
}

static __device__ __forceinline__ v8f wmma16x16x32(v16h a, v16h b, v8f c) {
  // D(f32 16x16) = A(f16 16x32) * B(f16 32x16) + C
  return __builtin_amdgcn_wmma_f32_16x16x32_f16(false, a, false, b, (short)0, c,
                                                false, false);
}

// Async memory->LDS copy, 16B per lane, tracked by ASYNCcnt (ISA 08 §4).
// Generic shared pointers truncate to the LDS byte offset (ISA §10.2).
static __device__ __forceinline__ void async_ld_b128(void* lds, const void* g) {
  unsigned lo = (unsigned)(size_t)lds;
  unsigned long long ga = (unsigned long long)(size_t)g;
  asm volatile("global_load_async_to_lds_b128 %0, %1, off"
               :
               : "v"(lo), "v"(ga)
               : "memory");
}
static __device__ __forceinline__ void wait_async0() {
  asm volatile("s_wait_asynccnt 0" ::: "memory");
}

// ---------------------------------------------------------------------------
// 1) fp32 -> fp16 conversion; weights are stored transposed: Wt[n][k] = W[k][n]
// ---------------------------------------------------------------------------
__global__ void mhsa_cvt_kernel(const float* __restrict__ x,
                                const float* __restrict__ Wq,
                                const float* __restrict__ Wk,
                                const float* __restrict__ Wv,
                                const float* __restrict__ Wo,
                                half_t* __restrict__ xh,
                                half_t* __restrict__ Wqt,
                                half_t* __restrict__ Wkt,
                                half_t* __restrict__ Wvt,
                                half_t* __restrict__ Wot) {
  const int NX = N_TOK * C_DIM;   // 2^21
  const int NW = C_DIM * C_DIM;   // 2^18
  int tid = blockIdx.x * blockDim.x + threadIdx.x;
  if (tid < NX) {
    xh[tid] = (half_t)x[tid];
  } else {
    int j = tid - NX;
    int w = j >> 18;
    int i = j & (NW - 1);
    int n = i >> 9, k = i & (C_DIM - 1);
    const float* src;
    half_t* dst;
    switch (w) {
      case 0: src = Wq; dst = Wqt; break;
      case 1: src = Wk; dst = Wkt; break;
      case 2: src = Wv; dst = Wvt; break;
      default: src = Wo; dst = Wot; break;
    }
    dst[i] = (half_t)src[k * C_DIM + n];
  }
}

// ---------------------------------------------------------------------------
// 2) Shared WMMA GEMM mainloop: Y[4096,512] = A[4096,512] @ W (+bias in caller)
//    Block: 256 threads = 8 waves; each wave owns 32 M-rows x 64 N-cols.
//    LDS tiles are double-buffered and filled with async memory->LDS DMA
//    (ASYNCcnt): the next tile streams in while wmma runs on the current one.
// ---------------------------------------------------------------------------
#define BM 256
#define BN 64
#define BK 32
#define NKSTEP (C_DIM / BK)  // 16
#define LDA 40
#define LDB 40

static __device__ __forceinline__ void gemm_core(
    const half_t* __restrict__ A, const half_t* __restrict__ Bt,
    half_t* __restrict__ AsBase, half_t* __restrict__ BsBase, int row0,
    int col0, v8f (&acc)[2][4]) {
  const int tid = threadIdx.x;
  const int wid = tid >> 5;
  const int lane = tid & 31;
  const int ml = lane & 15;
  const int hh = lane >> 4;

  // per-thread staging addresses
  const half_t* gA = A + (size_t)(row0 + tid) * C_DIM;  // 1 row / thread
  const int bn = tid >> 2, bk = (tid & 3) * 8;
  const half_t* gB = Bt + (size_t)(col0 + bn) * C_DIM + bk;

  auto stage = [&](int buf, int k0) {
    const half_t* g = gA + k0;
    half_t* as = AsBase + buf * (BM * LDA);
    half_t* bs = BsBase + buf * (BN * LDB);
#pragma unroll
    for (int c = 0; c < 4; ++c)
      async_ld_b128(&as[tid * LDA + c * 8], g + c * 8);
    async_ld_b128(&bs[bn * LDB + bk], gB + k0);
  };

  stage(0, 0);
  for (int kk = 0; kk < NKSTEP; ++kk) {
    wait_async0();      // our tile-kk DMA done
    __syncthreads();    // everyone's done (and prior readers of next buf done)
    if (kk + 1 < NKSTEP) {
      stage((kk + 1) & 1, (kk + 1) * BK);   // overlap DMA with wmma below
      if (kk + 2 < NKSTEP) {                // warm GL2 one tile further out
        __builtin_prefetch(gA + (kk + 2) * BK, 0, 1);
        __builtin_prefetch(gB + (kk + 2) * BK, 0, 1);
      }
    }
    const half_t* as = AsBase + (kk & 1) * (BM * LDA);
    const half_t* bs = BsBase + (kk & 1) * (BN * LDB);

    // A fragments: lane row m, k = 8h+0..7 (v0-3) and 16+8h+0..7 (v4-7)
    const int m0 = wid * 32;
    v16h a0 = hcat(*(const v8h*)&as[(m0 + ml) * LDA + 8 * hh],
                   *(const v8h*)&as[(m0 + ml) * LDA + 16 + 8 * hh]);
    v16h a1 = hcat(*(const v8h*)&as[(m0 + 16 + ml) * LDA + 8 * hh],
                   *(const v8h*)&as[(m0 + 16 + ml) * LDA + 16 + 8 * hh]);
    v16h b[4];
#pragma unroll
    for (int nt = 0; nt < 4; ++nt) {
      const half_t* bp = &bs[(nt * 16 + ml) * LDB + 8 * hh];
      b[nt] = hcat(*(const v8h*)bp, *(const v8h*)(bp + 16));
    }
#pragma unroll
    for (int nt = 0; nt < 4; ++nt) {
      acc[0][nt] = wmma16x16x32(a0, b[nt], acc[0][nt]);
      acc[1][nt] = wmma16x16x32(a1, b[nt], acc[1][nt]);
    }
  }
}

// Fused Q/K/V projections: gridDim.z selects the weight/bias/output.
//   z=0: Q -> [H][N][D] f16, scaled by 1/sqrt(C)
//   z=1: K -> [H][N][D] f16
//   z=2: V -> [H][D][N] f16 (transposed per head)
__global__ __launch_bounds__(256) void mhsa_gemm_qkv_kernel(
    const half_t* __restrict__ xh, const half_t* __restrict__ Wqt,
    const half_t* __restrict__ Wkt, const half_t* __restrict__ Wvt,
    const float* __restrict__ bq, const float* __restrict__ bk,
    const float* __restrict__ bv, half_t* __restrict__ Qs,
    half_t* __restrict__ Kh, half_t* __restrict__ Vt) {
  __shared__ alignas(16) half_t As[2][BM * LDA];
  __shared__ alignas(16) half_t Bs[2][BN * LDB];

  const int z = blockIdx.z;
  const half_t* Bt = (z == 0) ? Wqt : (z == 1) ? Wkt : Wvt;
  const float* bias = (z == 0) ? bq : (z == 1) ? bk : bv;
  const int row0 = blockIdx.y * BM;
  const int col0 = blockIdx.x * BN;

  v8f acc[2][4] = {};
  gemm_core(xh, Bt, &As[0][0], &Bs[0][0], row0, col0, acc);

  const int tid = threadIdx.x;
  const int wid = tid >> 5;
  const int ml = tid & 15;
  const int hh = (tid & 31) >> 4;
  const float qscale = 0.044194173824159216f;  // 1/sqrt(512)
#pragma unroll
  for (int mt = 0; mt < 2; ++mt) {
#pragma unroll
    for (int nt = 0; nt < 4; ++nt) {
      int c = col0 + nt * 16 + ml;
      float bval = bias[c];
      int hd = c >> 6, d = c & 63;
#pragma unroll
      for (int v = 0; v < 8; ++v) {
        int r = row0 + wid * 32 + mt * 16 + v + 8 * hh;  // C layout: m=v+8*hh
        float val = acc[mt][nt][v] + bval;
        if (z == 0) {
          Qs[((size_t)hd * N_TOK + r) * D_HEAD + d] = (half_t)(val * qscale);
        } else if (z == 1) {
          Kh[((size_t)hd * N_TOK + r) * D_HEAD + d] = (half_t)val;
        } else {
          Vt[((size_t)hd * D_HEAD + d) * N_TOK + r] = (half_t)val;
        }
      }
    }
  }
}

// Output projection: out(fp32)[4096,512] = Oh @ Wo + bo
__global__ __launch_bounds__(256) void mhsa_gemm_out_kernel(
    const half_t* __restrict__ Oh, const half_t* __restrict__ Wot,
    const float* __restrict__ bo, float* __restrict__ out) {
  __shared__ alignas(16) half_t As[2][BM * LDA];
  __shared__ alignas(16) half_t Bs[2][BN * LDB];

  const int row0 = blockIdx.y * BM;
  const int col0 = blockIdx.x * BN;

  v8f acc[2][4] = {};
  gemm_core(Oh, Wot, &As[0][0], &Bs[0][0], row0, col0, acc);

  const int tid = threadIdx.x;
  const int wid = tid >> 5;
  const int ml = tid & 15;
  const int hh = (tid & 31) >> 4;
#pragma unroll
  for (int mt = 0; mt < 2; ++mt) {
#pragma unroll
    for (int nt = 0; nt < 4; ++nt) {
      int c = col0 + nt * 16 + ml;
      float bval = bo[c];
#pragma unroll
      for (int v = 0; v < 8; ++v) {
        int r = row0 + wid * 32 + mt * 16 + v + 8 * hh;
        out[(size_t)r * C_DIM + c] = acc[mt][nt][v] + bval;
      }
    }
  }
}

// ---------------------------------------------------------------------------
// 3) Flash attention: grid (32 q-tiles, 8 heads), 8 waves x 16 query rows.
//    K/V tiles double-buffered via async memory->LDS DMA (ASYNCcnt);
//    per 32-key step: S = Qs @ K^T (4 wmma), online softmax, O += P @ V (4 wmma).
// ---------------------------------------------------------------------------
#define KT 32
#define NITER (N_TOK / KT)  // 128
#define LDK 72
#define LDV 40
#define LDP 40

__global__ __launch_bounds__(256) void mhsa_flash_kernel(
    const half_t* __restrict__ Q,   // [H][N][D], pre-scaled
    const half_t* __restrict__ K,   // [H][N][D]
    const half_t* __restrict__ Vt,  // [H][D][N]
    half_t* __restrict__ O) {       // [N][C]
  __shared__ alignas(16) half_t Ks[2][KT * LDK];
  __shared__ alignas(16) half_t Vs[2][D_HEAD * LDV];
  __shared__ alignas(16) half_t Ps[8][16 * LDP];

  const int tid = threadIdx.x;
  const int wid = tid >> 5;
  const int lane = tid & 31;
  const int ml = lane & 15;
  const int hh = lane >> 4;
  const int head = blockIdx.y;
  const int q0 = blockIdx.x * 128;

  // persistent Q fragments for this wave's 16 rows (two 16x32 k-chunks of D=64)
  const half_t* qp = Q + ((size_t)head * N_TOK + q0 + wid * 16 + ml) * D_HEAD;
  v16h aq0 = hcat(*(const v8h*)(qp + 8 * hh), *(const v8h*)(qp + 16 + 8 * hh));
  v16h aq1 = hcat(*(const v8h*)(qp + 32 + 8 * hh), *(const v8h*)(qp + 48 + 8 * hh));

  // per-thread staging addresses (16B each for K and V tiles)
  const int skey = tid >> 3, sd8 = (tid & 7) * 8;
  const half_t* gK = K + ((size_t)head * N_TOK + skey) * D_HEAD + sd8;
  const int sdd = tid >> 2, sk8 = (tid & 3) * 8;
  const half_t* gV = Vt + ((size_t)head * D_HEAD + sdd) * N_TOK + sk8;

  auto stage = [&](int buf, int k0) {
    async_ld_b128(&Ks[buf][skey * LDK + sd8], gK + (size_t)k0 * D_HEAD);
    async_ld_b128(&Vs[buf][sdd * LDV + sk8], gV + k0);
  };

  v8f o0 = {}, o1 = {}, o2 = {}, o3 = {};
  float rM[8], rL[8];
#pragma unroll
  for (int v = 0; v < 8; ++v) { rM[v] = -3.0e38f; rL[v] = 0.f; }

  stage(0, 0);
  for (int it = 0; it < NITER; ++it) {
    const int k0 = it * KT;
    wait_async0();
    __syncthreads();
    if (it + 1 < NITER) {
      stage((it + 1) & 1, k0 + KT);  // overlap next K/V DMA with compute
      __builtin_prefetch(gK + (size_t)(k0 + 2 * KT) * D_HEAD, 0, 1);
      __builtin_prefetch(gV + k0 + 2 * KT, 0, 1);
    }
    const half_t* ks = &Ks[it & 1][0];
    const half_t* vs = &Vs[it & 1][0];

    // S = Qs @ K^T : two 16x16 score tiles (keys 0..15 and 16..31)
    v8f c0 = {}, c1 = {};
    {
      const half_t* kp0 = &ks[ml * LDK];
      const half_t* kp1 = &ks[(16 + ml) * LDK];
      v16h b00 = hcat(*(const v8h*)(kp0 + 8 * hh), *(const v8h*)(kp0 + 16 + 8 * hh));
      v16h b01 = hcat(*(const v8h*)(kp0 + 32 + 8 * hh), *(const v8h*)(kp0 + 48 + 8 * hh));
      v16h b10 = hcat(*(const v8h*)(kp1 + 8 * hh), *(const v8h*)(kp1 + 16 + 8 * hh));
      v16h b11 = hcat(*(const v8h*)(kp1 + 32 + 8 * hh), *(const v8h*)(kp1 + 48 + 8 * hh));
      c0 = wmma16x16x32(aq0, b00, c0);
      c0 = wmma16x16x32(aq1, b01, c0);
      c1 = wmma16x16x32(aq0, b10, c1);
      c1 = wmma16x16x32(aq1, b11, c1);
    }

    // online softmax: row stats live per accumulator index v (row m = v + 8*hh)
    half_t* pw = &Ps[wid][0];
#pragma unroll
    for (int v = 0; v < 8; ++v) {
      float x = fmaxf(c0[v], c1[v]);
      x = fmaxf(x, __shfl_xor(x, 1, 32));
      x = fmaxf(x, __shfl_xor(x, 2, 32));
      x = fmaxf(x, __shfl_xor(x, 4, 32));
      x = fmaxf(x, __shfl_xor(x, 8, 32));
      float mnew = fmaxf(rM[v], x);
      float alpha = __expf(rM[v] - mnew);
      rM[v] = mnew;
      float p0 = __expf(c0[v] - mnew);
      float p1 = __expf(c1[v] - mnew);
      float s = p0 + p1;
      s += __shfl_xor(s, 1, 32);
      s += __shfl_xor(s, 2, 32);
      s += __shfl_xor(s, 4, 32);
      s += __shfl_xor(s, 8, 32);
      rL[v] = rL[v] * alpha + s;
      o0[v] *= alpha; o1[v] *= alpha; o2[v] *= alpha; o3[v] *= alpha;
      int m = v + 8 * hh;
      pw[m * LDP + ml] = (half_t)p0;       // keys 0..15
      pw[m * LDP + 16 + ml] = (half_t)p1;  // keys 16..31
    }
    // wave-private LDS round trip (C-layout -> A-fragment layout); LDS ops are
    // in-order per wave, the wait covers cross-lane visibility.
    asm volatile("s_wait_dscnt 0" ::: "memory");
    v16h pa = hcat(*(const v8h*)&pw[ml * LDP + 8 * hh],
                   *(const v8h*)&pw[ml * LDP + 16 + 8 * hh]);
    v16h vb[4];
#pragma unroll
    for (int t = 0; t < 4; ++t) {
      const half_t* vp = &vs[(t * 16 + ml) * LDV + 8 * hh];
      vb[t] = hcat(*(const v8h*)vp, *(const v8h*)(vp + 16));
    }
    o0 = wmma16x16x32(pa, vb[0], o0);
    o1 = wmma16x16x32(pa, vb[1], o1);
    o2 = wmma16x16x32(pa, vb[2], o2);
    o3 = wmma16x16x32(pa, vb[3], o3);
  }

  // normalize and store O as f16 [N][C]
#pragma unroll
  for (int v = 0; v < 8; ++v) {
    float inv = 1.0f / rL[v];
    int r = q0 + wid * 16 + v + 8 * hh;
    half_t* orow = O + (size_t)r * C_DIM + head * D_HEAD;
    orow[0 + ml]  = (half_t)(o0[v] * inv);
    orow[16 + ml] = (half_t)(o1[v] * inv);
    orow[32 + ml] = (half_t)(o2[v] * inv);
    orow[48 + ml] = (half_t)(o3[v] * inv);
  }
}

// ---------------------------------------------------------------------------
extern "C" void kernel_launch(void* const* d_in, const int* in_sizes, int n_in,
                              void* d_out, int out_size, void* d_ws,
                              size_t ws_size, hipStream_t stream) {
  const float* x  = (const float*)d_in[0];
  const float* Wq = (const float*)d_in[1];
  const float* bq = (const float*)d_in[2];
  const float* Wk = (const float*)d_in[3];
  const float* bk = (const float*)d_in[4];
  const float* Wv = (const float*)d_in[5];
  const float* bv = (const float*)d_in[6];
  const float* Wo = (const float*)d_in[7];
  const float* bo = (const float*)d_in[8];
  float* out = (float*)d_out;

  char* ws = (char*)d_ws;
  size_t off = 0;
  auto alloc = [&](size_t bytes) -> void* {
    void* p = ws + off;
    off = (off + bytes + 255) & ~(size_t)255;
    return p;
  };
  half_t* xh  = (half_t*)alloc((size_t)N_TOK * C_DIM * 2);
  half_t* Wqt = (half_t*)alloc((size_t)C_DIM * C_DIM * 2);
  half_t* Wkt = (half_t*)alloc((size_t)C_DIM * C_DIM * 2);
  half_t* Wvt = (half_t*)alloc((size_t)C_DIM * C_DIM * 2);
  half_t* Wot = (half_t*)alloc((size_t)C_DIM * C_DIM * 2);
  half_t* Qs  = (half_t*)alloc((size_t)H_NUM * N_TOK * D_HEAD * 2);
  half_t* Kh  = (half_t*)alloc((size_t)H_NUM * N_TOK * D_HEAD * 2);
  half_t* Vt  = (half_t*)alloc((size_t)H_NUM * D_HEAD * N_TOK * 2);
  half_t* Oh  = (half_t*)alloc((size_t)N_TOK * C_DIM * 2);

  // 1) convert x + transpose-convert weights
  {
    int total = N_TOK * C_DIM + 4 * C_DIM * C_DIM;  // 3,145,728
    mhsa_cvt_kernel<<<total / 256, 256, 0, stream>>>(x, Wq, Wk, Wv, Wo, xh, Wqt,
                                                     Wkt, Wvt, Wot);
  }

  // 2) fused Q/K/V projections (z selects target), 384 workgroups
  mhsa_gemm_qkv_kernel<<<dim3(C_DIM / BN, N_TOK / BM, 3), 256, 0, stream>>>(
      xh, Wqt, Wkt, Wvt, bq, bk, bv, Qs, Kh, Vt);

  // 3) fused attention
  mhsa_flash_kernel<<<dim3(N_TOK / 128, H_NUM), 256, 0, stream>>>(Qs, Kh, Vt, Oh);

  // 4) output projection -> fp32 d_out
  mhsa_gemm_out_kernel<<<dim3(C_DIM / BN, N_TOK / BM), 256, 0, stream>>>(
      Oh, Wot, bo, out);
}
